// ResNetBasicblock_34711925686310
// MI455X (gfx1250) — 8-bit matrix operand order repaired and checked by reading, compiled, not run on hardware
//
#include <hip/hip_runtime.h>

typedef __attribute__((ext_vector_type(8))) int v8i;
typedef __attribute__((ext_vector_type(4))) int v4i;

#define CIN   256
#define COUT  256
#define HH    56
#define WW    56
#define BATCH 32
#define HWSZ  (HH * WW)
#define NPIX  (BATCH * HWSZ)
#define KTOT  (CIN * 9)
#define PADH  58
#define XB_BYTES ((size_t)BATCH * PADH * PADH * CIN)
#define WA_BYTES ((size_t)COUT * KTOT)

typedef __attribute__((address_space(1))) v4i glb_v4i;
typedef __attribute__((address_space(3))) v4i lds_v4i;

__device__ __forceinline__ void async_copy_b128(void* lds, const void* g) {
#if __has_builtin(__builtin_amdgcn_global_load_async_to_lds_b128)
  __builtin_amdgcn_global_load_async_to_lds_b128(
      (glb_v4i*)(size_t)g, (lds_v4i*)(unsigned)(size_t)lds, 0, 0);
#else
  unsigned lo = (unsigned)(size_t)lds;
  unsigned long long ga = (unsigned long long)(size_t)g;
  asm volatile("global_load_async_to_lds_b128 %0, %1, off"
               :: "v"(lo), "v"(ga) : "memory");
#endif
}

__device__ __forceinline__ void wait_async0() {
#if __has_builtin(__builtin_amdgcn_s_wait_asynccnt)
  __builtin_amdgcn_s_wait_asynccnt(0);
#else
  asm volatile("s_wait_asynccnt 0" ::: "memory");
#endif
}

__global__ __launch_bounds__(256) void prep_weights(const float* __restrict__ w,
                                                    char* __restrict__ wA) {
  int i = blockIdx.x * 256 + threadIdx.x;
  int cout = i / KTOT;
  int r = i - cout * KTOT;
  int tap = r >> 8;
  int cin = r & 255;
  float v = w[(cout * CIN + cin) * 9 + tap];
  wA[i] = (v > -1e-6f) ? (char)1 : (char)-1;
}

__global__ __launch_bounds__(256) void sign_pack_x(const float* __restrict__ x,
                                                   char* __restrict__ xb) {
  int i = blockIdx.x * 256 + threadIdx.x;
  int bc = i / HWSZ;
  int hw = i - bc * HWSZ;
  int b = bc >> 8;
  int c = bc & 255;
  int h = hw / WW;
  int w = hw - h * WW;
  float v = x[i];
  xb[((size_t)((b * PADH + h + 1) * PADH + (w + 1)) * CIN) + c] =
      (v > -1e-6f) ? (char)1 : (char)-1;
}

__device__ __forceinline__ void load_tiles_async(const char* __restrict__ A,
                                                 const char* __restrict__ X,
                                                 char* as, char* bs,
                                                 const int* off0,
                                                 int m0, int ks, int t) {
  int tap = ks >> 2;
  int c0  = (ks & 3) << 6;
  int kh  = tap / 3, kw = tap - kh * 3;
  int tapoff = (kh * PADH + kw) * CIN;
  int row  = t >> 1;
  int half = t & 1;
  const char* ga = A + (size_t)(m0 + row) * KTOT + tap * CIN + c0 + half * 32;
  const char* gb = X + (size_t)off0[row] + tapoff + c0 + half * 32;
  char* la = as + row * 64 + half * 32;
  char* lb = bs + row * 64 + half * 32;
  async_copy_b128(la,      ga);
  async_copy_b128(la + 16, ga + 16);
  async_copy_b128(lb,      gb);
  async_copy_b128(lb + 16, gb + 16);
}

__device__ __forceinline__ void compute_step(const char* as, const char* bs,
                                             v8i acc[2][4],
                                             int wm, int wn, int lane) {
  int hi = lane >> 4;
  int ml = lane & 15;
  v8i a[2], b[4];
  #pragma unroll
  for (int mt = 0; mt < 2; ++mt) {
    const char* p = as + (wm * 32 + mt * 16 + ml) * 64 + hi * 16;
    #pragma unroll
    for (int tq = 0; tq < 4; ++tq) {
      const int* q = (const int*)(p + (tq & 1) * 8 + (tq >> 1) * 32);
      a[mt][2 * tq]     = q[0];
      a[mt][2 * tq + 1] = q[1];
    }
  }
  #pragma unroll
  for (int nt = 0; nt < 4; ++nt) {
    const char* p = bs + (wn * 64 + nt * 16 + ml) * 64 + hi * 16;
    v4i lo = *(const v4i*)p;
    v4i up = *(const v4i*)(p + 32);
    #pragma unroll
    for (int j = 0; j < 4; ++j) { b[nt][j] = lo[j]; b[nt][4 + j] = up[j]; }
  }
  #pragma unroll
  for (int mt = 0; mt < 2; ++mt)
    #pragma unroll
    for (int nt = 0; nt < 4; ++nt)
      acc[mt][nt] = __builtin_amdgcn_wmma_i32_16x16x64_iu8(
          true, a[mt], true, b[nt], acc[mt][nt], false, false);
}

__global__ __launch_bounds__(256, 2) void binconv_gemm(const char* __restrict__ A,
                                                       const char* __restrict__ X,
                                                       float* __restrict__ Y) {
  __shared__ char As[2][128 * 64];
  __shared__ char Bs[2][128 * 64];
  __shared__ int  off0[128];

  const int t    = threadIdx.x;
  const int lane = t & 31;
  const int wave = t >> 5;
  const int wm   = wave & 3;
  const int wn   = wave >> 2;
  const int m0   = blockIdx.y * 128;
  const int n0   = blockIdx.x * 128;

  if (t < 128) {
    int p  = n0 + t;
    int b  = p / HWSZ;
    int hw = p - b * HWSZ;
    int h  = hw / WW;
    int w  = hw - h * WW;
    off0[t] = ((b * PADH + h) * PADH + w) * CIN;
  }
  __syncthreads();

  v8i zero = {};
  v8i acc[2][4];
  #pragma unroll
  for (int i = 0; i < 2; ++i)
    #pragma unroll
    for (int j = 0; j < 4; ++j) acc[i][j] = zero;

  load_tiles_async(A, X, As[0], Bs[0], off0, m0, 0, t);

  int cur = 0;
  for (int ks = 0; ks < 36; ++ks) {
    wait_async0();
    __syncthreads();
    if (ks + 1 < 36)
      load_tiles_async(A, X, As[cur ^ 1], Bs[cur ^ 1], off0, m0, ks + 1, t);
    compute_step(As[cur], Bs[cur], acc, wm, wn, lane);
    cur ^= 1;
  }

  int hi = lane >> 4;
  int ml = lane & 15;
  #pragma unroll
  for (int mt = 0; mt < 2; ++mt) {
    #pragma unroll
    for (int nt = 0; nt < 4; ++nt) {
      int pix = n0 + wn * 64 + nt * 16 + ml;
      #pragma unroll
      for (int r = 0; r < 8; ++r) {
        int cout = m0 + wm * 32 + mt * 16 + hi * 8 + r;
        Y[(size_t)cout * NPIX + pix] = (float)acc[mt][nt][r];
      }
    }
  }
}

__global__ __launch_bounds__(256) void bn_stats(const float* __restrict__ Y,
                                                float* __restrict__ sums) {
  __shared__ float rs[256], rq[256];
  int c = blockIdx.x;
  size_t base = (size_t)c * NPIX + blockIdx.y * 2048 + threadIdx.x;
  float s = 0.f, q = 0.f;
  #pragma unroll
  for (int i = 0; i < 8; ++i) {
    float v = Y[base + i * 256];
    s += v;
    q += v * v;
  }
  rs[threadIdx.x] = s;
  rq[threadIdx.x] = q;
  __syncthreads();
  for (int off = 128; off > 0; off >>= 1) {
    if (threadIdx.x < (unsigned)off) {
      rs[threadIdx.x] += rs[threadIdx.x + off];
      rq[threadIdx.x] += rq[threadIdx.x + off];
    }
    __syncthreads();
  }
  if (threadIdx.x == 0) {
    atomicAdd(&sums[c], rs[0]);
    atomicAdd(&sums[256 + c], rq[0]);
  }
}

__global__ void bn_finalize(const float* __restrict__ sums,
                            const float* __restrict__ gamma,
                            const float* __restrict__ beta,
                            float* __restrict__ ss) {
  int c = threadIdx.x;
  float inv  = 1.0f / (float)NPIX;
  float mean = sums[c] * inv;
  float var  = sums[256 + c] * inv - mean * mean;
  float sc   = gamma[c] * rsqrtf(var + 1e-5f);
  ss[c]       = sc;
  ss[256 + c] = beta[c] - mean * sc;
}

__global__ __launch_bounds__(256) void bn_binarize(const float* __restrict__ Y,
                                                   const float* __restrict__ ss,
                                                   char* __restrict__ xb) {
  int p = blockIdx.x;
  int c = threadIdx.x;
  float v = Y[(size_t)c * NPIX + p] * ss[c] + ss[256 + c];
  int b  = p / HWSZ;
  int hw = p - b * HWSZ;
  int h  = hw / WW;
  int w  = hw - h * WW;
  xb[((size_t)((b * PADH + h + 1) * PADH + (w + 1)) * CIN) + c] =
      (v > -1e-6f) ? (char)1 : (char)-1;
}

__global__ __launch_bounds__(256) void final_add(const float* __restrict__ x,
                                                 const float* __restrict__ Y,
                                                 const float* __restrict__ ss,
                                                 float* __restrict__ out) {
  int i = blockIdx.x * 256 + threadIdx.x;
  int bc = i / HWSZ;
  int hw = i - bc * HWSZ;
  int b = bc >> 8;
  int c = bc & 255;
  out[i] = x[i] + Y[(size_t)c * NPIX + b * HWSZ + hw] * ss[c] + ss[256 + c];
}

extern "C" void kernel_launch(void* const* d_in, const int* in_sizes, int n_in,
                              void* d_out, int out_size, void* d_ws, size_t ws_size,
                              hipStream_t stream) {
  const float* x       = (const float*)d_in[0];
  const float* w_a     = (const float*)d_in[1];
  const float* gamma_a = (const float*)d_in[2];
  const float* beta_a  = (const float*)d_in[3];
  const float* w_b     = (const float*)d_in[4];
  const float* gamma_b = (const float*)d_in[5];
  const float* beta_b  = (const float*)d_in[6];
  float* out = (float*)d_out;

  char* ws = (char*)d_ws;
  size_t o = 0;
  char* xb  = ws + o; o += XB_BYTES;
  char* wA1 = ws + o; o += WA_BYTES;
  char* wA2 = ws + o; o += WA_BYTES;
  o = (o + 255) & ~(size_t)255;
  float* y    = (float*)(ws + o); o += (size_t)COUT * NPIX * 4;
  float* sums = (float*)(ws + o); o += 2048;
  float* ss1  = (float*)(ws + o); o += 2048;
  float* ss2  = (float*)(ws + o); o += 2048;

  (void)hipMemsetAsync(xb, 0, XB_BYTES, stream);
  prep_weights<<<(int)(WA_BYTES / 256), 256, 0, stream>>>(w_a, wA1);
  prep_weights<<<(int)(WA_BYTES / 256), 256, 0, stream>>>(w_b, wA2);
  sign_pack_x<<<(BATCH * CIN * HWSZ) / 256, 256, 0, stream>>>(x, xb);

  dim3 g(NPIX / 128, COUT / 128);
  binconv_gemm<<<g, 256, 0, stream>>>(wA1, xb, y);

  (void)hipMemsetAsync(sums, 0, 2048, stream);
  bn_stats<<<dim3(COUT, NPIX / 2048), 256, 0, stream>>>(y, sums);
  bn_finalize<<<1, 256, 0, stream>>>(sums, gamma_a, beta_a, ss1);
  bn_binarize<<<NPIX, 256, 0, stream>>>(y, ss1, xb);

  binconv_gemm<<<g, 256, 0, stream>>>(wA2, xb, y);

  (void)hipMemsetAsync(sums, 0, 2048, stream);
  bn_stats<<<dim3(COUT, NPIX / 2048), 256, 0, stream>>>(y, sums);
  bn_finalize<<<1, 256, 0, stream>>>(sums, gamma_b, beta_b, ss2);

  final_add<<<(BATCH * CIN * HWSZ) / 256, 256, 0, stream>>>(x, y, ss2, out);
}
